// Model_38062000177324
// MI455X (gfx1250) — matrix operand order repaired and checked by reading, not compiled, not run on hardware
//
#include <hip/hip_runtime.h>
#include <hip/hip_bf16.h>
#include <math.h>

typedef __attribute__((ext_vector_type(16))) _Float16 v16h;
typedef __attribute__((ext_vector_type(4)))  _Float16 v4h;
typedef __attribute__((ext_vector_type(8)))  float    v8f;

namespace lstm_cfg {
constexpr int F_IN = 13;
constexpr int H    = 128;
constexpr int B    = 256;
constexpr int T    = 512;
constexpr int NG   = 4 * H;      // 512 gate outputs
constexpr int ROWS = 16;         // batch rows per workgroup
constexpr int KTOT = 160;        // state = [h(128) | x(13, padded to 32)]
constexpr int NCHUNK = KTOT / 32;// 5 K-chunks of 32

// LDS layout (bytes)
constexpr int WCAT_OFF  = 0;                            // [512][160] f16 = 163840
constexpr int SBUF_OFF  = WCAT_OFF + NG * KTOT * 2;     // [16][160] f16  = 5120
constexpr int BIAS_OFF  = SBUF_OFF + ROWS * KTOT * 2;   // [512] f32      = 2048
constexpr int PROJW_OFF = BIAS_OFF + NG * 4;            // [13][128] f32  = 6656
constexpr int PROJB_OFF = PROJW_OFF + F_IN * H * 4;     // [13] f32
constexpr int LDS_BYTES = PROJB_OFF + F_IN * 4 + 64;    // ~173.7 KB (< 320 KB/WGP)
}

// Branchless activations: v_exp_f32 + v_rcp_f32 only (no libm control flow).
__device__ __forceinline__ float sigmoid_fast(float x) {
    return __builtin_amdgcn_rcpf(1.0f + __expf(-x));
}
__device__ __forceinline__ float tanh_fast(float x) {
    // tanh(x) = 1 - 2/(e^{2x} + 1); monotone, branch-free, saturates correctly
    return 1.0f - 2.0f * __builtin_amdgcn_rcpf(__expf(2.0f * x) + 1.0f);
}

__global__ __launch_bounds__(256, 1)
void lstm_autoenc_kernel(const float* __restrict__ X,
                         const float* __restrict__ W_ih,
                         const float* __restrict__ W_hh,
                         const float* __restrict__ b_ih,
                         const float* __restrict__ b_hh,
                         const float* __restrict__ projW,
                         const float* __restrict__ projb,
                         float* __restrict__ out)
{
    using namespace lstm_cfg;
    extern __shared__ char smem[];
    _Float16* wcat = reinterpret_cast<_Float16*>(smem + WCAT_OFF); // [512][160] = [W_hh | W_ih pad]
    _Float16* sbuf = reinterpret_cast<_Float16*>(smem + SBUF_OFF); // [16][160]  = [h | x pad]
    float*    bias = reinterpret_cast<float*>(smem + BIAS_OFF);    // b_ih + b_hh
    float*    pW   = reinterpret_cast<float*>(smem + PROJW_OFF);
    float*    pB   = reinterpret_cast<float*>(smem + PROJB_OFF);

    const int tid  = threadIdx.x;
    const int b0   = blockIdx.x * ROWS;   // 16 WGs cover B=256
    const int wv   = tid >> 5;            // wave 0..7 -> hidden slice [16*wv, 16*wv+16)
    const int lane = tid & 31;
    const int ncol = lane & 15;           // A row / B column / C column
    const int hlf  = lane >> 4;

    // ---------------- one-time init: stage concatenated weights to LDS (f16) ----------------
    for (int i = tid; i < NG * KTOT; i += 256) {
        int n = i / KTOT, k = i - n * KTOT;
        _Float16 w = (_Float16)0.f;
        if (k < H)                 w = (_Float16)W_hh[n * H + k];
        else if (k < H + F_IN)     w = (_Float16)W_ih[n * F_IN + (k - H)];
        wcat[i] = w;
    }
    for (int i = tid; i < NG; i += 256)       bias[i] = b_ih[i] + b_hh[i];
    for (int i = tid; i < F_IN * H; i += 256) pW[i]   = projW[i];
    if (tid < F_IN) pB[tid] = projb[tid];
    for (int i = tid; i < ROWS * KTOT; i += 256) sbuf[i] = (_Float16)0.f; // h=0, pad=0
    __syncthreads();

    // per-lane gate biases (C-layout: value depends only on column n)
    const float bG0 = bias[(wv     ) * 16 + ncol];  // i
    const float bG1 = bias[(wv + 8 ) * 16 + ncol];  // f
    const float bG2 = bias[(wv + 16) * 16 + ncol];  // g
    const float bG3 = bias[(wv + 24) * 16 + ncol];  // o

    // A fragment (16x32 f16): lane = row m, K = kb + (v<4?0:16) + 8*hlf + 2*(v&3)
    auto loadA = [&](const _Float16* base, int kb) {
        union { v16h h; unsigned u[8]; } fr;
#pragma unroll
        for (int v = 0; v < 8; ++v) {
            int k = kb + ((v < 4) ? 0 : 16) + hlf * 8 + ((v & 3) << 1);
            fr.u[v] = *reinterpret_cast<const unsigned*>(base + ncol * KTOT + k);
        }
        return fr.h;
    };
    // B fragment (32x16 f16): lane = column n, K = kb + (v<4?0:16) + 8*hlf + 2*(v&3)
    auto loadB = [&](const _Float16* base, int kb) {
        union { v16h h; unsigned u[8]; } fr;
#pragma unroll
        for (int v = 0; v < 8; ++v) {
            int k = kb + ((v < 4) ? 0 : 16) + hlf * 8 + ((v & 3) << 1);
            fr.u[v] = *reinterpret_cast<const unsigned*>(base + ncol * KTOT + k);
        }
        return fr.h;
    };
    auto mm = [&](v16h a, v16h b, v8f c) {
        return __builtin_amdgcn_wmma_f32_16x16x32_f16(false, a, false, b,
                                                      (short)0, c, false, false);
    };

    // ---- hoist ALL weight B-fragments into registers for the whole recurrence ----
    // wb[gate][chunk]; gate tiles: i -> wv, f -> wv+8, g -> wv+16, o -> wv+24
    v16h wb[4][NCHUNK];
#pragma unroll
    for (int q = 0; q < 4; ++q) {
        const _Float16* gb = wcat + (wv + 8 * q) * 16 * KTOT;
#pragma unroll
        for (int c = 0; c < NCHUNK; ++c)
            wb[q][c] = loadB(gb, 32 * c);
    }

    v8f cfrag, hfrag;
#pragma unroll
    for (int v = 0; v < 8; ++v) { cfrag[v] = 0.f; hfrag[v] = 0.f; }

    const int jcol = wv * 16 + ncol;

    // One LSTM cell step. Contains one internal barrier (after sbuf reads).
    auto cell = [&]() {
        // A fragments: full state [h | x] for this WG's 16 batch rows
        v16h a[NCHUNK];
#pragma unroll
        for (int c = 0; c < NCHUNK; ++c) a[c] = loadA(sbuf, 32 * c);
        __syncthreads();   // all sbuf reads retired before sbuf is rewritten

        v8f g0, g1, g2, g3;
#pragma unroll
        for (int v = 0; v < 8; ++v) { g0[v] = bG0; g1[v] = bG1; g2[v] = bG2; g3[v] = bG3; }

        // 20 back-to-back WMMAs, 4 independent accumulator chains interleaved
#pragma unroll
        for (int c = 0; c < NCHUNK; ++c) {
            g0 = mm(a[c], wb[0][c], g0);
            g1 = mm(a[c], wb[1][c], g1);
            g2 = mm(a[c], wb[2][c], g2);
            g3 = mm(a[c], wb[3][c], g3);
        }

        // i/f/g/o for this wave's hidden slice are all local -> c stays in registers
#pragma unroll
        for (int v = 0; v < 8; ++v) {
            float iv = sigmoid_fast(g0[v]);
            float fv = sigmoid_fast(g1[v]);
            float gv = tanh_fast(g2[v]);
            float ov = sigmoid_fast(g3[v]);
            float cn = fv * cfrag[v] + iv * gv;
            cfrag[v] = cn;
            hfrag[v] = ov * tanh_fast(cn);
        }
#pragma unroll
        for (int v = 0; v < 8; ++v)
            sbuf[(v + 8 * hlf) * KTOT + jcol] = (_Float16)hfrag[v];
    };

    // ---------------- encoder: t = 0..T-1 (x load software-pipelined 1 step ahead) ----------
    const bool stager = (tid < ROWS * F_IN);
    const int  sm = tid / F_IN;
    const int  sf = tid - sm * F_IN;
    const float* xcol = X + (size_t)(b0 + sm) * T * F_IN + sf;  // step stride = F_IN

    float xreg = stager ? xcol[0] : 0.0f;   // preload t=0
    for (int t = 0; t < T; ++t) {
        if (stager) {
            sbuf[sm * KTOT + H + sf] = (_Float16)xreg;
            if (t + 1 < T) xreg = xcol[(size_t)(t + 1) * F_IN];  // issue now, wait next step
        }
        __syncthreads();   // x staged; previous step's h writes visible
        cell();            // internal barrier inside
    }

    // embeddings = h_n : out[0 .. B*H)
#pragma unroll
    for (int v = 0; v < 8; ++v)
        out[(size_t)(b0 + v + 8 * hlf) * H + jcol] = hfrag[v];

    // seed decoder input with X[:, 0, :]
    if (stager)
        sbuf[sm * KTOT + H + sf] = (_Float16)xcol[0];
    __syncthreads();

    // ---------------- decoder: autoregressive, t = 0..T-1 ----------------
    float* yout = out + (size_t)B * H;
    for (int t = 0; t < T; ++t) {
        cell();            // internal barrier inside
        __syncthreads();   // new h visible to projection phase
        if (stager) {
            const _Float16* hr = sbuf + sm * KTOT;   // h row = cols 0..127
            const float*    wr = pW + sf * H;
            float acc = pB[sf];
#pragma unroll
            for (int j4 = 0; j4 < H / 4; ++j4) {
                v4h hv = *reinterpret_cast<const v4h*>(hr + 4 * j4);
                acc += (float)hv[0] * wr[4 * j4 + 0] + (float)hv[1] * wr[4 * j4 + 1]
                     + (float)hv[2] * wr[4 * j4 + 2] + (float)hv[3] * wr[4 * j4 + 3];
            }
            yout[((size_t)(b0 + sm) * T + t) * F_IN + sf] = acc;
            sbuf[sm * KTOT + H + sf] = (_Float16)acc;   // next decoder input
        }
        __syncthreads();   // x(t+1) staged before next cell reads it
    }
}

extern "C" void kernel_launch(void* const* d_in, const int* in_sizes, int n_in,
                              void* d_out, int out_size, void* d_ws, size_t ws_size,
                              hipStream_t stream) {
    using namespace lstm_cfg;
    const float* X     = (const float*)d_in[0];
    const float* W_ih  = (const float*)d_in[1];
    const float* W_hh  = (const float*)d_in[2];
    const float* b_ih  = (const float*)d_in[3];
    const float* b_hh  = (const float*)d_in[4];
    const float* projW = (const float*)d_in[5];
    const float* projb = (const float*)d_in[6];
    float* out = (float*)d_out;

    hipLaunchKernelGGL(lstm_autoenc_kernel, dim3(B / ROWS), dim3(256), LDS_BYTES, stream,
                       X, W_ih, W_hh, b_ih, b_hh, projW, projb, out);
}
